// BinaryMLP_90881507983459
// MI455X (gfx1250) — 8-bit matrix operand order repaired and checked by reading, compiled, not run on hardware
//
#include <hip/hip_runtime.h>
#include <stdint.h>

typedef __attribute__((ext_vector_type(16))) int   v16i;
typedef __attribute__((ext_vector_type(8)))  float v8f;

#define M_TILE   64
#define NWAVES   8
#define THREADS  256
#define H        512
#define KP1      896
#define NBATCH   16384

#define FP8_P1   0x38
#define FP8_M1   0xB8

__device__ __forceinline__ v16i load_a128(const uint8_t* p) {
  const int2* q = (const int2*)p;
  v16i a;
#pragma unroll
  for (int i = 0; i < 8; ++i) {
    int2 c = q[4 * (i >> 1) + (i & 1)];
    a[2 * i] = c.x; a[2 * i + 1] = c.y;
  }
  return a;
}
__device__ __forceinline__ v16i load_b128(const uint8_t* p) {
  const int4* q = (const int4*)p;
  v16i b;
#pragma unroll
  for (int i = 0; i < 4; ++i) {
    int4 c = q[2 * i];
    b[4 * i] = c.x; b[4 * i + 1] = c.y; b[4 * i + 2] = c.z; b[4 * i + 3] = c.w;
  }
  return b;
}

__device__ __forceinline__ v8f wmma_fp8(v16i a, v16i b, v8f c) {
  return __builtin_amdgcn_wmma_f32_16x16x128_fp8_fp8(a, b, (short)0, c, false, false);
}

template <int K>
__device__ __forceinline__ void hidden_layer(
    const uint8_t* __restrict__ actIn,
    const uint8_t* __restrict__ W,
    const float* __restrict__ bi, const float* __restrict__ g,
    const float* __restrict__ be, const float* __restrict__ mu,
    const float* __restrict__ vr, float eps,
    uint8_t* __restrict__ actOut,
    int lane, int wave)
{
  const int hi = (lane >> 4) * 8;
  const int ln = lane & 15;
#pragma unroll 1
  for (int u = wave; u < (M_TILE / 32) * (H / 32); u += NWAVES) {
    const int mtp = u >> 4;
    const int ntp = u & 15;
    const uint8_t* a0b = actIn + (size_t)(mtp * 32 + ln) * K + 2 * hi;
    const uint8_t* a1b = a0b + (size_t)16 * K;
    const uint8_t* b0b = W + (size_t)(ntp * 32 + ln) * K + 2 * hi;
    const uint8_t* b1b = b0b + (size_t)16 * K;

    v8f acc00 = {}, acc01 = {}, acc10 = {}, acc11 = {};
#pragma unroll 1
    for (int kb = 0; kb < K; kb += 128) {
      v16i a0 = load_a128(a0b + kb);
      v16i a1 = load_a128(a1b + kb);
      v16i b0 = load_b128(b0b + kb);
      v16i b1 = load_b128(b1b + kb);
      acc00 = wmma_fp8(a0, b0, acc00);
      acc01 = wmma_fp8(a0, b1, acc01);
      acc10 = wmma_fp8(a1, b0, acc10);
      acc11 = wmma_fp8(a1, b1, acc11);
    }

    auto epi = [&](int sm, int sn, const v8f& acc) {
      const int col = ntp * 32 + sn * 16 + ln;
      const float scale = g[col] * rsqrtf(vr[col] + eps);
      const float bia = bi[col], m = mu[col], beta = be[col];
#pragma unroll
      for (int r = 0; r < 8; ++r) {
        const int row = mtp * 32 + sm * 16 + r + hi;
        const float y  = acc[r] + bia;
        const float bn = (y - m) * scale + beta;
        actOut[row * H + col] = (bn >= 0.0f) ? (uint8_t)FP8_P1 : (uint8_t)FP8_M1;
      }
    };
    epi(0, 0, acc00); epi(0, 1, acc01); epi(1, 0, acc10); epi(1, 1, acc11);
  }
}

__global__ __launch_bounds__(THREADS, 4)
void bmlp_fused(const uint8_t* __restrict__ xb,
                const uint8_t* __restrict__ w1b,
                const uint8_t* __restrict__ w2b,
                const uint8_t* __restrict__ w3b,
                const uint8_t* __restrict__ w4b,
                const float* __restrict__ b1, const float* __restrict__ g1,
                const float* __restrict__ be1, const float* __restrict__ m1,
                const float* __restrict__ v1,
                const float* __restrict__ b2, const float* __restrict__ g2,
                const float* __restrict__ be2, const float* __restrict__ m2,
                const float* __restrict__ v2,
                const float* __restrict__ b3, const float* __restrict__ g3,
                const float* __restrict__ be3, const float* __restrict__ m3,
                const float* __restrict__ v3,
                const float* __restrict__ b4,
                float* __restrict__ out)
{
  extern __shared__ uint8_t smem[];
  uint8_t* actA = smem;
  uint8_t* actB = smem + M_TILE * H;

  const int lane = threadIdx.x & 31;
  const int wave = threadIdx.x >> 5;
  const size_t row0 = (size_t)blockIdx.x * M_TILE;

  hidden_layer<KP1>(xb + row0 * KP1, w1b, b1, g1, be1, m1, v1, 1e-5f, actA, lane, wave);
  __syncthreads();
  hidden_layer<H>(actA, w2b, b2, g2, be2, m2, v2, 1e-5f, actB, lane, wave);
  __syncthreads();
  hidden_layer<H>(actB, w3b, b3, g3, be3, m3, v3, 512.0f, actA, lane, wave);
  __syncthreads();

  {
    const int hi = (lane >> 4) * 8;
    const int ln = lane & 15;
#pragma unroll 1
    for (int t = wave; t < M_TILE / 16; t += NWAVES) {
      const uint8_t* abase = actA + (size_t)(t * 16 + ln) * H + 2 * hi;
      const uint8_t* bbase = w4b  + (size_t)ln * H + 2 * hi;
      v8f acc = {};
#pragma unroll 1
      for (int kb = 0; kb < H; kb += 128) {
        v16i a = load_a128(abase + kb);
        v16i b = load_b128(bbase + kb);
        acc = wmma_fp8(a, b, acc);
      }
      if (ln < 10) {
        const float bia = b4[ln];
#pragma unroll
        for (int r = 0; r < 8; ++r) {
          const size_t grow = row0 + t * 16 + r + hi;
          out[grow * 10 + ln] = acc[r] + bia;
        }
      }
    }
  }
}

__global__ void binarize_fp8_pad(const float* __restrict__ src, uint8_t* __restrict__ dst,
                                 int src_rows, int src_cols, int dst_cols, long total)
{
  long i = (long)blockIdx.x * blockDim.x + threadIdx.x;
  if (i >= total) return;
  const int r = (int)(i / dst_cols);
  const int c = (int)(i % dst_cols);
  uint8_t v = 0;
  if (r < src_rows && c < src_cols)
    v = (src[(size_t)r * src_cols + c] >= 0.0f) ? (uint8_t)FP8_P1 : (uint8_t)FP8_M1;
  dst[i] = v;
}

extern "C" void kernel_launch(void* const* d_in, const int* in_sizes, int n_in,
                              void* d_out, int out_size, void* d_ws, size_t ws_size,
                              hipStream_t stream) {
  (void)in_sizes; (void)n_in; (void)out_size; (void)ws_size;
  const float* x   = (const float*)d_in[0];
  const float* w1  = (const float*)d_in[1];
  const float* b1  = (const float*)d_in[2];
  const float* g1  = (const float*)d_in[3];
  const float* be1 = (const float*)d_in[4];
  const float* m1  = (const float*)d_in[5];
  const float* v1  = (const float*)d_in[6];
  const float* w2  = (const float*)d_in[7];
  const float* b2  = (const float*)d_in[8];
  const float* g2  = (const float*)d_in[9];
  const float* be2 = (const float*)d_in[10];
  const float* m2  = (const float*)d_in[11];
  const float* v2  = (const float*)d_in[12];
  const float* w3  = (const float*)d_in[13];
  const float* b3  = (const float*)d_in[14];
  const float* g3  = (const float*)d_in[15];
  const float* be3 = (const float*)d_in[16];
  const float* m3  = (const float*)d_in[17];
  const float* v3  = (const float*)d_in[18];
  const float* w4  = (const float*)d_in[19];
  const float* b4  = (const float*)d_in[20];
  float* out = (float*)d_out;

  uint8_t* ws  = (uint8_t*)d_ws;
  uint8_t* xb  = ws;
  uint8_t* w1b = xb  + (size_t)NBATCH * KP1;
  uint8_t* w2b = w1b + (size_t)H * KP1;
  uint8_t* w3b = w2b + (size_t)H * H;
  uint8_t* w4b = w3b + (size_t)H * H;

  auto bin = [&](const float* s, uint8_t* d, int sr, int sc, int dr, int dc) {
    const long total = (long)dr * dc;
    const int blocks = (int)((total + 255) / 256);
    binarize_fp8_pad<<<blocks, 256, 0, stream>>>(s, d, sr, sc, dc, total);
  };
  bin(x,  xb,  NBATCH, 784, NBATCH, KP1);
  bin(w1, w1b, H,      784, H,      KP1);
  bin(w2, w2b, H,      H,   H,      H);
  bin(w3, w3b, H,      H,   H,      H);
  bin(w4, w4b, 10,     H,   16,     H);

  const dim3 grid(NBATCH / M_TILE);
  const size_t shmem = (size_t)2 * M_TILE * H;
  bmlp_fused<<<grid, THREADS, shmem, stream>>>(
      xb, w1b, w2b, w3b, w4b,
      b1, g1, be1, m1, v1,
      b2, g2, be2, m2, v2,
      b3, g3, be3, m3, v3,
      b4, out);
}
